// LocalSelfAttention_44762149159259
// MI455X (gfx1250) — matrix operand order repaired and checked by reading, not compiled, not run on hardware
//
#include <hip/hip_runtime.h>

// ---------- types ----------
typedef __attribute__((ext_vector_type(16))) __bf16 v16bf;
typedef __attribute__((ext_vector_type(8)))  float  v8f;
typedef __attribute__((ext_vector_type(4)))  unsigned int uint4v;

// A-fragment (16x32 bf16, ISA 7.12.2): lane holds K = {base..base+7, base+16..base+23}
__device__ __forceinline__ v16bf load_frag_a(const __bf16* p) {
  union { uint4v u[2]; v16bf v; } t;
  t.u[0] = *(const uint4v*)(p);
  t.u[1] = *(const uint4v*)(p + 16);
  return t.v;
}
// B-fragment (32x16 bf16): lane holds K = {base..base+7, base+16..base+23}
__device__ __forceinline__ v16bf load_frag_b(const __bf16* p) {
  union { uint4v u[2]; v16bf v; } t;
  t.u[0] = *(const uint4v*)(p);
  t.u[1] = *(const uint4v*)(p + 16);
  return t.v;
}
__device__ __forceinline__ v8f wmma_bf16(v16bf a, v16bf b, v8f c) {
  return __builtin_amdgcn_wmma_f32_16x16x32_bf16(false, a, false, b, (short)0, c,
                                                 false, false);
}

// async global->LDS copy of 16 bytes per lane (CDNA5 async path, ASYNCcnt)
__device__ __forceinline__ void async_copy16(const __bf16* gsrc, __bf16* ldst) {
  unsigned long long ga = (unsigned long long)gsrc;
  unsigned la = (unsigned)(unsigned long long)ldst;   // low 32 bits = LDS offset
  asm volatile("global_load_async_to_lds_b128 %0, %1, off"
               :: "v"(la), "v"(ga) : "memory");
}

// ---------- problem constants ----------
#define BB   4
#define SS   4096
#define HH   1024
#define NHH  16
#define DHH  64
#define CLL  128
#define NCC  32
#define QKV_ELEMS 16777216ull   // B*S*NH*DH

// =====================================================================
// Kernel 0a: X f32 -> bf16 (same [16384,1024] layout)
// =====================================================================
__global__ __launch_bounds__(256) void cvt_x_kernel(
    const float* __restrict__ X, __bf16* __restrict__ Xb)
{
  size_t i8 = ((size_t)blockIdx.x * 256 + threadIdx.x) * 8;
  float4 a = *(const float4*)(X + i8);
  float4 b = *(const float4*)(X + i8 + 4);
  union { __bf16 h[8]; uint4v u; } o;
  o.h[0] = (__bf16)a.x; o.h[1] = (__bf16)a.y;
  o.h[2] = (__bf16)a.z; o.h[3] = (__bf16)a.w;
  o.h[4] = (__bf16)b.x; o.h[5] = (__bf16)b.y;
  o.h[6] = (__bf16)b.z; o.h[7] = (__bf16)b.w;
  *(uint4v*)(Xb + i8) = o.u;
}

// =====================================================================
// Kernel 0b: W f32 [k][n] -> bf16 transposed Wt [n][k] (per z in q,k,v)
// =====================================================================
__global__ __launch_bounds__(256) void cvt_w_kernel(
    const float* __restrict__ Wq, const float* __restrict__ Wk,
    const float* __restrict__ Wv, __bf16* __restrict__ WtBase)
{
  __shared__ __align__(16) __bf16 T[64 * 72];
  const int z = blockIdx.z;
  const float* W = (z == 0) ? Wq : ((z == 1) ? Wk : Wv);
  __bf16* Wt = WtBase + (size_t)z * (1024u * 1024u);
  const int k0 = blockIdx.x * 64;
  const int n0 = blockIdx.y * 64;
  const int tid = threadIdx.x;

  for (int i = 0; i < 4; ++i) {
    int idx = tid + i * 256;          // 0..1023
    int r = idx >> 4;                 // k row 0..63
    int q = idx & 15;                 // n quad 0..15
    float4 v = *(const float4*)(W + (size_t)(k0 + r) * 1024 + n0 + q * 4);
    T[(q * 4 + 0) * 72 + r] = (__bf16)v.x;
    T[(q * 4 + 1) * 72 + r] = (__bf16)v.y;
    T[(q * 4 + 2) * 72 + r] = (__bf16)v.z;
    T[(q * 4 + 3) * 72 + r] = (__bf16)v.w;
  }
  __syncthreads();
  for (int i = 0; i < 2; ++i) {
    int idx = tid + i * 256;          // 0..511
    int n = idx >> 3;                 // 0..63
    int o8 = (idx & 7) * 8;
    *(uint4v*)(Wt + (size_t)(n0 + n) * 1024 + k0 + o8) =
        *(const uint4v*)&T[n * 72 + o8];
  }
}

// =====================================================================
// Kernel 1: QKV GEMM, async double-buffered LDS, 64x64 wave tiles
//   Xb[16384,1024]bf16 * Wt[z][n,k]bf16 -> [B,NH,S,DH] bf16 (+f32 acc)
// =====================================================================
__global__ __launch_bounds__(256) void qkv_gemm_kernel(
    const __bf16* __restrict__ Xb, const __bf16* __restrict__ WtBase,
    __bf16* __restrict__ qkvws)
{
  __shared__ __align__(16) __bf16 Abuf[2][256 * 72];   // [m][k], stride 72
  __shared__ __align__(16) __bf16 Bbuf[2][128 * 72];   // [n][k], stride 72

  const int z = blockIdx.z;
  const __bf16* Wt = WtBase + (size_t)z * (1024u * 1024u);
  const float scale = (z == 1) ? 0.125f : 1.0f;
  __bf16* out = qkvws + (size_t)z * QKV_ELEMS;

  const int m0 = blockIdx.y * 256;
  const int n0 = blockIdx.x * 128;
  const int tid  = threadIdx.x;
  const int lane = tid & 31;
  const int wave = tid >> 5;
  const int wm = (wave >> 1) * 64;   // 4 waves in M
  const int wn = (wave & 1) * 64;    // 2 waves in N
  const int l  = lane & 15;
  const int hi = lane >> 4;

  v8f acc[4][4];
  for (int mt = 0; mt < 4; ++mt)
    for (int nt = 0; nt < 4; ++nt)
      for (int j = 0; j < 8; ++j) acc[mt][nt][j] = 0.0f;

  // issue one 64-deep k-stage: 12 async instructions per wave
  auto issue_stage = [&](int k0, int buf) {
    for (int i = 0; i < 8; ++i) {
      int c = tid + i * 256;          // 0..2047 : 256 rows x 8 chunks
      int row = c >> 3;
      int o8  = (c & 7) * 8;
      async_copy16(Xb + (size_t)(m0 + row) * HH + k0 + o8,
                   &Abuf[buf][row * 72 + o8]);
    }
    for (int i = 0; i < 4; ++i) {
      int c = tid + i * 256;          // 0..1023 : 128 rows x 8 chunks
      int n  = c >> 3;
      int o8 = (c & 7) * 8;
      async_copy16(Wt + (size_t)(n0 + n) * 1024 + k0 + o8,
                   &Bbuf[buf][n * 72 + o8]);
    }
  };

  issue_stage(0, 0);
  int cur = 0;
  for (int k0 = 0; k0 < HH; k0 += 64) {
    if (k0 + 64 < HH) {
      issue_stage(k0 + 64, cur ^ 1);
      asm volatile("s_wait_asynccnt 12" ::: "memory");  // stage 'cur' landed
    } else {
      asm volatile("s_wait_asynccnt 0" ::: "memory");
    }
    __syncthreads();

    const __bf16* A  = Abuf[cur];
    const __bf16* Bs = Bbuf[cur];
    for (int kc = 0; kc < 64; kc += 32) {
      v16bf a[4], b[4];
      for (int mt = 0; mt < 4; ++mt)
        a[mt] = load_frag_a(&A[(wm + mt * 16 + l) * 72 + kc + hi * 8]);
      for (int nt = 0; nt < 4; ++nt)
        b[nt] = load_frag_b(&Bs[(wn + nt * 16 + l) * 72 + kc + hi * 8]);
      for (int mt = 0; mt < 4; ++mt)
        for (int nt = 0; nt < 4; ++nt)
          acc[mt][nt] = wmma_bf16(a[mt], b[nt], acc[mt][nt]);
    }
    __syncthreads();
    cur ^= 1;
  }

  // write out: [B,NH,S,DH] bf16
  for (int mt = 0; mt < 4; ++mt)
    for (int nt = 0; nt < 4; ++nt)
      for (int i = 0; i < 8; ++i) {
        int m = m0 + wm + mt * 16 + i + hi * 8;
        int c = n0 + wn + nt * 16 + l;
        int b  = m >> 12;
        int s  = m & 4095;
        int h  = c >> 6;
        int d  = c & 63;
        out[(((size_t)(b * NHH + h)) * SS + s) * DHH + d] =
            (__bf16)(acc[mt][nt][i] * scale);
      }
}

// =====================================================================
// Kernel 2: chunked local attention, flash-style online softmax
// =====================================================================
__global__ __launch_bounds__(256) void local_attn_kernel(
    const __bf16* __restrict__ q, const __bf16* __restrict__ k,
    const __bf16* __restrict__ v, float* __restrict__ out)
{
  __shared__ __align__(16) __bf16 Ks[256 * 72];        // [key][d], prev||cur
  __shared__ __align__(16) __bf16 Vt[64 * 264];        // [d][key] transposed
  __shared__ __align__(16) __bf16 Ps[8 * 16 * 136];    // per-wave P tile

  const int chunk = blockIdx.x & 31;
  const int h     = (blockIdx.x >> 5) & 15;
  const int b     = blockIdx.x >> 9;
  const int prev  = (chunk + NCC - 1) & 31;            // roll-wrap like reference
  const size_t headBase = ((size_t)(b * NHH + h)) * SS * DHH;

  const int tid  = threadIdx.x;
  const int lane = tid & 31;
  const int wave = tid >> 5;
  const int l    = lane & 15;
  const int hi   = lane >> 4;

  // ---- stage K via async copy; V transposed via VGPRs ----
  for (int i = 0; i < 8; ++i) {
    int e   = (tid + i * 256) * 8;
    int key = e >> 6;                   // 0..255
    int d0  = e & 63;
    int sc  = (key < 128) ? prev : chunk;
    int sk  = key & 127;
    size_t srcOff = headBase + ((size_t)sc * CLL + sk) * DHH + d0;
    async_copy16(k + srcOff, &Ks[key * 72 + d0]);
    uint4v vv = *(const uint4v*)(v + srcOff);
    const __bf16* pv = (const __bf16*)&vv;
    for (int j = 0; j < 8; ++j) Vt[(d0 + j) * 264 + key] = pv[j];
  }
  asm volatile("s_wait_asynccnt 0" ::: "memory");
  __syncthreads();

  const int qrow0 = wave * 16;
  const __bf16* qb = q + headBase + (size_t)chunk * CLL * DHH;
  v16bf aq[2];
  for (int kc = 0; kc < 2; ++kc)
    aq[kc] = load_frag_a(qb + (qrow0 + l) * DHH + kc * 32 + hi * 8);

  float mrow[8], lrow[8];
  v8f acc[4];
  for (int i = 0; i < 8; ++i) { mrow[i] = -1e30f; lrow[i] = 0.0f; }
  for (int dt = 0; dt < 4; ++dt)
    for (int j = 0; j < 8; ++j) acc[dt][j] = 0.0f;

  __bf16* Pw = &Ps[wave * 16 * 136];

  for (int kb = 0; kb < 2; ++kb) {
    // S = Q K^T over this 128-key block
    v8f st[8];
    for (int t = 0; t < 8; ++t)
      for (int j = 0; j < 8; ++j) st[t][j] = 0.0f;
    for (int t = 0; t < 8; ++t)
      for (int kc = 0; kc < 2; ++kc) {
        v16bf bk = load_frag_b(&Ks[(kb * 128 + t * 16 + l) * 72 + kc * 32 + hi * 8]);
        st[t] = wmma_bf16(aq[kc], bk, st[t]);
      }

    // mask: kb==0 -> prev chunk (visible iff chunk>0); kb==1 -> causal diag
    for (int t = 0; t < 8; ++t) {
      int col = t * 16 + l;
      for (int i = 0; i < 8; ++i) {
        int row = qrow0 + i + hi * 8;
        bool vis = (kb == 0) ? (chunk > 0) : (row >= col);
        if (!vis) st[t][i] = -1e9f;
      }
    }

    // online softmax (rows split across 16-lane halves)
    for (int i = 0; i < 8; ++i) {
      float mx = st[0][i];
      for (int t = 1; t < 8; ++t) mx = fmaxf(mx, st[t][i]);
      for (int off = 1; off < 16; off <<= 1) mx = fmaxf(mx, __shfl_xor(mx, off, 32));
      float mnew = fmaxf(mrow[i], mx);
      float rsum = 0.0f;
      for (int t = 0; t < 8; ++t) {
        float p = __expf(st[t][i] - mnew);
        st[t][i] = p;
        rsum += p;
      }
      for (int off = 1; off < 16; off <<= 1) rsum += __shfl_xor(rsum, off, 32);
      float corr = __expf(mrow[i] - mnew);
      lrow[i] = lrow[i] * corr + rsum;
      mrow[i] = mnew;
      for (int dt = 0; dt < 4; ++dt) acc[dt][i] = acc[dt][i] * corr;
    }

    // P -> per-wave LDS -> A-fragments
    for (int t = 0; t < 8; ++t)
      for (int i = 0; i < 8; ++i)
        Pw[(i + hi * 8) * 136 + t * 16 + l] = (__bf16)st[t][i];

    for (int kc = 0; kc < 4; ++kc) {
      v16bf ap = load_frag_a(&Pw[l * 136 + kc * 32 + hi * 8]);
      for (int dt = 0; dt < 4; ++dt) {
        v16bf bv = load_frag_b(&Vt[(dt * 16 + l) * 264 + kb * 128 + kc * 32 + hi * 8]);
        acc[dt] = wmma_bf16(ap, bv, acc[dt]);
      }
    }
  }

  // normalize and write f32 output [B, S, NH*DH]
  for (int dt = 0; dt < 4; ++dt)
    for (int i = 0; i < 8; ++i) {
      float o = acc[dt][i] / lrow[i];
      int row = chunk * CLL + qrow0 + i + hi * 8;
      int col = h * DHH + dt * 16 + l;
      out[((size_t)b * SS + row) * (NHH * DHH) + col] = o;
    }
}

// =====================================================================
extern "C" void kernel_launch(void* const* d_in, const int* in_sizes, int n_in,
                              void* d_out, int out_size, void* d_ws, size_t ws_size,
                              hipStream_t stream) {
  const float* X  = (const float*)d_in[0];
  const float* Wq = (const float*)d_in[1];
  const float* Wk = (const float*)d_in[2];
  const float* Wv = (const float*)d_in[3];
  float* out = (float*)d_out;

  char* wsB = (char*)d_ws;
  __bf16* Xb  = (__bf16*)wsB;                                     // 16M elems
  __bf16* Wt  = (__bf16*)(wsB + QKV_ELEMS * 2);                   // 3 x 1M elems
  __bf16* qws = (__bf16*)(wsB + (QKV_ELEMS + 3ull * 1048576ull) * 2);
  __bf16* kws = qws + QKV_ELEMS;
  __bf16* vws = kws + QKV_ELEMS;

  cvt_x_kernel<<<dim3(8192), 256, 0, stream>>>(X, Xb);
  cvt_w_kernel<<<dim3(16, 16, 3), 256, 0, stream>>>(Wq, Wk, Wv, Wt);

  dim3 g1(8, 64, 3);    // N tiles, M tiles (256-high), {q,k,v}
  qkv_gemm_kernel<<<g1, 256, 0, stream>>>(Xb, Wt, qws);

  dim3 g2(BB * NHH * NCC);   // 2048
  local_attn_kernel<<<g2, 256, 0, stream>>>(qws, kws, vws, out);
}
